// MultiHeadAttention_61942018343359
// MI455X (gfx1250) — matrix operand order repaired and checked by reading, not compiled, not run on hardware
//
#include <hip/hip_runtime.h>

typedef _Float16 half_t;
typedef __attribute__((ext_vector_type(16))) _Float16 v16h;
typedef __attribute__((ext_vector_type(8)))  _Float16 v8h;
typedef __attribute__((ext_vector_type(8)))  float    v8f;

constexpr int cB  = 2;
constexpr int cS  = 2048;
constexpr int cD  = 1024;
constexpr int cH  = 16;
constexpr int cHD = 64;
constexpr int cM  = cB * cS;              // 4096 rows
constexpr float cSCALE = 0.125f;          // 64^-0.5

// Assemble a 16x32 f16 A-fragment for v_wmma_f32_16x16x32_f16.
// Layout used here: lanes 0-15 hold K=[0..15] ; lanes 16-31 hold K=[16..31] (contiguous, same as B)
// caller passes the two 8-half chunk addresses (already offset by hi*16 / hi*16+8).
__device__ __forceinline__ v16h make_a_frag(const half_t* p0, const half_t* p1) {
    v8h lo = *(const v8h*)p0;
    v8h hh = *(const v8h*)p1;
    v16h a;
#pragma unroll
    for (int i = 0; i < 8; ++i) { a[i] = lo[i]; a[i + 8] = hh[i]; }
    return a;
}

__device__ __forceinline__ v8f wmma_f16(v16h a, v16h b, v8f c) {
    return __builtin_amdgcn_wmma_f32_16x16x32_f16(false, a, false, b, (short)0, c, false, false);
}

// ---------------- conversion kernels ----------------

__global__ void cvt_f32_to_f16(const float* __restrict__ in, half_t* __restrict__ out, int n) {
    int i = blockIdx.x * blockDim.x + threadIdx.x;
    int stride = gridDim.x * blockDim.x;
    for (; i < n; i += stride) out[i] = (half_t)in[i];
}

// W[h][d][e] (H,D,HD f32) -> Wt[h][e][d] (f16) so B-fragments read contiguous K(=d).
__global__ void cvt_weight_tr(const float* __restrict__ W, half_t* __restrict__ Wt) {
    int i = blockIdx.x * blockDim.x + threadIdx.x;
    int stride = gridDim.x * blockDim.x;
    const int n = cH * cD * cHD;
    for (; i < n; i += stride) {
        int h = i / (cD * cHD);
        int r = i - h * (cD * cHD);
        int d = r / cHD;
        int e = r - d * cHD;
        Wt[(h * cHD + e) * cD + d] = (half_t)W[i];
    }
}

// ---------------- register-blocked WMMA GEMM ----------------
// out[m][n] = sum_k A[m][k]*Bt[n][k] + bias[n]
// A:  [cM, cD] f16 row-major.  Bt: [cD, cD] f16 row-major, row index = output column.
// Each wave computes a 32x64 tile: 2 M-subtiles x 4 N-subtiles -> 8 WMMAs per K-step
// from 2 A-fragment + 4 B-fragment loads (bytes/WMMA cut ~2.7x vs 16x16 tiling).
// MODE 0: f16 out [cM, cD]
// MODE 1: f16 out transposed per head: [B][H][HD][S]   (V^T for attention)
// MODE 2: f32 out [cM, cD]                              (final result)
template <int MODE>
__global__ void __launch_bounds__(128)
gemm_wmma(const half_t* __restrict__ A, const half_t* __restrict__ Bt,
          const float* __restrict__ bias, half_t* __restrict__ outH,
          float* __restrict__ outF) {
    const int lane = threadIdx.x & 31;
    const int wave = threadIdx.x >> 5;
    const int l    = lane & 15;
    const int hi   = lane >> 4;
    const int mBase = blockIdx.x * 32;                 // 2 subtiles of 16 rows
    const int nBase = (blockIdx.y * 4 + wave) * 64;    // 4 subtiles of 16 cols

    const half_t* aRow0 = A + (mBase + l) * cD;
    const half_t* aRow1 = A + (mBase + 16 + l) * cD;
    const half_t* bRow0 = Bt + (nBase +  0 + l) * cD + hi * 16;
    const half_t* bRow1 = Bt + (nBase + 16 + l) * cD + hi * 16;
    const half_t* bRow2 = Bt + (nBase + 32 + l) * cD + hi * 16;
    const half_t* bRow3 = Bt + (nBase + 48 + l) * cD + hi * 16;

    v8f c00 = {}, c01 = {}, c02 = {}, c03 = {};
    v8f c10 = {}, c11 = {}, c12 = {}, c13 = {};

    for (int k = 0; k < cD; k += 32) {
        v16h a0 = make_a_frag(aRow0 + k + hi * 16, aRow0 + k + hi * 16 + 8);
        v16h a1 = make_a_frag(aRow1 + k + hi * 16, aRow1 + k + hi * 16 + 8);
        v16h b0 = *(const v16h*)(bRow0 + k);
        v16h b1 = *(const v16h*)(bRow1 + k);
        v16h b2 = *(const v16h*)(bRow2 + k);
        v16h b3 = *(const v16h*)(bRow3 + k);
        c00 = wmma_f16(a0, b0, c00);
        c10 = wmma_f16(a1, b0, c10);
        c01 = wmma_f16(a0, b1, c01);
        c11 = wmma_f16(a1, b1, c11);
        c02 = wmma_f16(a0, b2, c02);
        c12 = wmma_f16(a1, b2, c12);
        c03 = wmma_f16(a0, b3, c03);
        c13 = wmma_f16(a1, b3, c13);
    }

    v8f cc[2][4] = {{c00, c01, c02, c03}, {c10, c11, c12, c13}};
#pragma unroll
    for (int mi = 0; mi < 2; ++mi) {
#pragma unroll
        for (int nb = 0; nb < 4; ++nb) {
            const int   col = nBase + nb * 16 + l;
            const float bv  = bias[col];
#pragma unroll
            for (int r = 0; r < 8; ++r) {
                const int   row = mBase + mi * 16 + r + hi * 8;  // C: VGPR r -> M = r+8*hi
                const float v   = cc[mi][nb][r] + bv;
                if constexpr (MODE == 0) {
                    outH[row * cD + col] = (half_t)v;
                } else if constexpr (MODE == 1) {
                    const int bb = row >> 11;            // / cS
                    const int s  = row & (cS - 1);
                    const int hh = col >> 6;             // / cHD
                    const int e  = col & (cHD - 1);
                    outH[((bb * cH + hh) * cHD + e) * cS + s] = (half_t)v;
                } else {
                    outF[row * cD + col] = v;
                }
            }
        }
    }
}

// ---------------- flash attention (1 wave per 16-query tile per head) ----------------
// Qp, Kp: [cM, cD] f16 (per-head slice via base offset, ld = cD)
// VTp:    [B][H][HD][S] f16 (V transposed so PV B-fragments are contiguous)
// Oc:     [cM, cD] f16  concat output (col = h*64 + e)
__global__ void __launch_bounds__(32)
attn_wmma(const half_t* __restrict__ Qp, const half_t* __restrict__ Kp,
          const half_t* __restrict__ VTp, half_t* __restrict__ Oc) {
    const int lane  = threadIdx.x & 31;
    const int l     = lane & 15;
    const int hi    = lane >> 4;
    const int qBase = blockIdx.x * 16;
    const int h     = blockIdx.y;
    const int b     = blockIdx.z;

    const half_t* Qbh  = Qp  + b * cS * cD + h * cHD;         // row stride cD
    const half_t* Kbh  = Kp  + b * cS * cD + h * cHD;
    const half_t* VTbh = VTp + ((b * cH + h) * cHD) * cS;     // [e][s], row stride cS

    __shared__ __align__(32) half_t Plds[16 * 32];

    // Preload Q A-fragments: HD=64 split into two K=32 chunks
    const half_t* qRow = Qbh + (qBase + l) * cD;
    const v16h aQ0 = make_a_frag(qRow +  0 + hi * 16, qRow +  8 + hi * 16);
    const v16h aQ1 = make_a_frag(qRow + 32 + hi * 16, qRow + 40 + hi * 16);

    float mstate[8], lstate[8];
    v8f acc[4];
#pragma unroll
    for (int r = 0; r < 8; ++r) { mstate[r] = -1e30f; lstate[r] = 0.0f; }
#pragma unroll
    for (int nb = 0; nb < 4; ++nb) acc[nb] = v8f{};

    for (int kt = 0; kt < cS; kt += 32) {
        // ---- scores: 16 queries x 32 keys (two 16-key C tiles, K=64 contraction) ----
        v8f sc0 = {}, sc1 = {};
        {
            const half_t* kRow0 = Kbh + (kt + l) * cD + hi * 16;
            const half_t* kRow1 = Kbh + (kt + 16 + l) * cD + hi * 16;
            v16h b00 = *(const v16h*)(kRow0);
            v16h b01 = *(const v16h*)(kRow0 + 32);
            v16h b10 = *(const v16h*)(kRow1);
            v16h b11 = *(const v16h*)(kRow1 + 32);
            sc0 = wmma_f16(aQ0, b00, sc0);
            sc0 = wmma_f16(aQ1, b01, sc0);
            sc1 = wmma_f16(aQ0, b10, sc1);
            sc1 = wmma_f16(aQ1, b11, sc1);
        }

        // ---- online softmax: rows live across 16-lane half-groups ----
        float p0[8], p1[8];
#pragma unroll
        for (int r = 0; r < 8; ++r) {
            float s0 = sc0[r] * cSCALE;
            float s1 = sc1[r] * cSCALE;
            float rm = fmaxf(s0, s1);
#pragma unroll
            for (int m = 1; m < 16; m <<= 1) rm = fmaxf(rm, __shfl_xor(rm, m, 32));
            const float mn = fmaxf(mstate[r], rm);
            const float sf = __expf(mstate[r] - mn);
            mstate[r] = mn;
            const float e0 = __expf(s0 - mn);
            const float e1 = __expf(s1 - mn);
            float ps = e0 + e1;
#pragma unroll
            for (int m = 1; m < 16; m <<= 1) ps += __shfl_xor(ps, m, 32);
            lstate[r] = lstate[r] * sf + ps;
            p0[r] = e0; p1[r] = e1;
#pragma unroll
            for (int nb = 0; nb < 4; ++nb) acc[nb][r] *= sf;
        }

        // ---- reshape P: C-layout f32 -> LDS (row-major 16x32 f16) -> A-fragment ----
#pragma unroll
        for (int r = 0; r < 8; ++r) {
            Plds[(r + hi * 8) * 32 + l]      = (half_t)p0[r];
            Plds[(r + hi * 8) * 32 + 16 + l] = (half_t)p1[r];
        }
        __syncthreads();
        v16h aP = make_a_frag(&Plds[l * 32 + hi * 16], &Plds[l * 32 + hi * 16 + 8]);

        // ---- P @ V : 4 output column blocks over HD=64 ----
#pragma unroll
        for (int nb = 0; nb < 4; ++nb) {
            v16h bv = *(const v16h*)(VTbh + (nb * 16 + l) * cS + kt + hi * 16);
            acc[nb] = wmma_f16(aP, bv, acc[nb]);
        }
        __syncthreads();
    }

    // ---- normalize and store concat output [b*S+s][h*64+e] ----
#pragma unroll
    for (int r = 0; r < 8; ++r) {
        const float inv = 1.0f / lstate[r];
        const int   row = b * cS + qBase + r + hi * 8;
#pragma unroll
        for (int nb = 0; nb < 4; ++nb) {
            const int col = h * cHD + nb * 16 + l;
            Oc[row * cD + col] = (half_t)(acc[nb][r] * inv);
        }
    }
}

// ---------------- launch ----------------

extern "C" void kernel_launch(void* const* d_in, const int* in_sizes, int n_in,
                              void* d_out, int out_size, void* d_ws, size_t ws_size,
                              hipStream_t stream) {
    const float* K_in = (const float*)d_in[0];
    const float* V_in = (const float*)d_in[1];
    const float* Q_in = (const float*)d_in[2];
    const float* Wk   = (const float*)d_in[3];
    const float* bk   = (const float*)d_in[4];
    const float* Wq   = (const float*)d_in[5];
    const float* bq   = (const float*)d_in[6];
    const float* Wv   = (const float*)d_in[7];
    const float* bv   = (const float*)d_in[8];
    const float* Wp   = (const float*)d_in[9];
    const float* bp   = (const float*)d_in[10];

    half_t* ws = (half_t*)d_ws;
    const size_t SZ_MD = (size_t)cM * cD;   // 4096*1024 halves
    const size_t SZ_W  = (size_t)cD * cD;   // 1024*1024 halves (H*HD*D == D*D)
    half_t* Xk  = ws; ws += SZ_MD;
    half_t* Xv  = ws; ws += SZ_MD;
    half_t* Xq  = ws; ws += SZ_MD;
    half_t* WkT = ws; ws += SZ_W;
    half_t* WqT = ws; ws += SZ_W;
    half_t* WvT = ws; ws += SZ_W;
    half_t* WpH = ws; ws += SZ_W;
    half_t* Qp  = ws; ws += SZ_MD;
    half_t* Kp  = ws; ws += SZ_MD;
    half_t* VTp = ws; ws += SZ_MD;
    half_t* Cc  = ws; ws += SZ_MD;

    const dim3 cb(256);
    cvt_f32_to_f16<<<2048, cb, 0, stream>>>(K_in, Xk, (int)SZ_MD);
    cvt_f32_to_f16<<<2048, cb, 0, stream>>>(V_in, Xv, (int)SZ_MD);
    cvt_f32_to_f16<<<2048, cb, 0, stream>>>(Q_in, Xq, (int)SZ_MD);
    cvt_f32_to_f16<<<1024, cb, 0, stream>>>(Wp,   WpH, (int)SZ_W);
    cvt_weight_tr <<<1024, cb, 0, stream>>>(Wk, WkT);
    cvt_weight_tr <<<1024, cb, 0, stream>>>(Wq, WqT);
    cvt_weight_tr <<<1024, cb, 0, stream>>>(Wv, WvT);

    // 32x64 tile per wave, 4 waves/block -> block covers 32 rows x 256 cols
    const dim3 ggrid(cM / 32, cD / 256);    // 128 x 4
    const dim3 gblock(128);
    gemm_wmma<0><<<ggrid, gblock, 0, stream>>>(Xq, WqT, bq, Qp,  nullptr);
    gemm_wmma<0><<<ggrid, gblock, 0, stream>>>(Xk, WkT, bk, Kp,  nullptr);
    gemm_wmma<1><<<ggrid, gblock, 0, stream>>>(Xv, WvT, bv, VTp, nullptr);

    const dim3 agrid(cS / 16, cH, cB);      // 128 x 16 x 2 blocks, 1 wave each
    attn_wmma<<<agrid, dim3(32), 0, stream>>>(Qp, Kp, VTp, Cc);

    gemm_wmma<2><<<ggrid, gblock, 0, stream>>>(Cc, WpH, bp, nullptr, (float*)d_out);
}
